// TransformerDecoderLayerRelative_120259085157
// MI455X (gfx1250) — compile-verified, matrix operand order checked by reading, not run on hardware
//
#include <hip/hip_runtime.h>

typedef __bf16 bf16;
typedef __attribute__((ext_vector_type(16))) __bf16 v16bf;
typedef __attribute__((ext_vector_type(8)))  float  v8f;

// Problem constants (match reference)
constexpr int Bc = 2, Sc = 1024, Mc = 1024, Dc = 1024, DIc = 1024;
constexpr int Hc = 16, DHc = 64, DFFc = 4096, Stc = Sc + Mc; // St = 2048

// ---------------------------------------------------------------------------
// WMMA helpers
// ---------------------------------------------------------------------------
union FragU { v16bf v; uint4 q[2]; };

__device__ __forceinline__ v16bf load_frag_bf16(const bf16* __restrict__ base,
                                                int ld, int lane) {
  // 16x32 bf16 tile, row-major with row stride ld (elements).
  // CDNA5 16-bit A layout: lane<16 -> row=lane, K in {0..7, 16..23};
  //                        lane>=16 -> row=lane-16, K in {8..15, 24..31}.
  int r  = lane & 15;
  int kh = (lane >> 4) << 3; // 0 or 8
  const bf16* p = base + (long)r * ld + kh;
  FragU f;
  f.q[0] = *reinterpret_cast<const uint4*>(p);
  f.q[1] = *reinterpret_cast<const uint4*>(p + 16);
  return f.v;
}

__device__ __forceinline__ v8f wmma_bf16(v16bf a, v16bf b, v8f c) {
  return __builtin_amdgcn_wmma_f32_16x16x32_bf16(false, a, false, b,
                                                 (short)0, c, false, false);
}

// ---------------------------------------------------------------------------
// NT GEMM: C[M,N] = A[M,K] * Bt[N,K]^T   (bf16 in, fp32 out)
// Block = 256 threads = 8 waves (4 M-waves x 2 N-waves).
// WN = wave N-tile (32 or 64): block tile = 128 x (2*WN).
// Wave tile 32xWN: 2 A-frags, WN/16 B-frags, 2*(WN/16) WMMAs per K=32 step.
// Batched over blockIdx.z with two-level (b,h) strides.
// M % 128 == 0, N % (2*WN) == 0, K % 32 == 0 guaranteed by caller.
// ---------------------------------------------------------------------------
template <int WN>
__global__ __launch_bounds__(256) void k_gemm_nt_bf16(
    const bf16* __restrict__ A, const bf16* __restrict__ Bt,
    float* __restrict__ C, int K, int lda, int ldb, int ldc,
    long sAb, long sAh, long sBb, long sBh, long sCb, long sCh, int Hdiv) {
  constexpr int NT = WN / 16; // B tiles per wave
  int z = blockIdx.z;
  int bb = z / Hdiv, hh = z % Hdiv;
  A  += bb * sAb + hh * sAh;
  Bt += bb * sBb + hh * sBh;
  C  += bb * sCb + hh * sCh;

  int tid = threadIdx.x, wave = tid >> 5, lane = tid & 31;
  int wm = wave >> 1, wn = wave & 1;
  int m0 = blockIdx.x * 128 + wm * 32;
  int n0 = blockIdx.y * (2 * WN) + wn * WN;

  const v8f vz = {0, 0, 0, 0, 0, 0, 0, 0};
  v8f acc[2][NT];
#pragma unroll
  for (int mi = 0; mi < 2; ++mi)
#pragma unroll
    for (int ni = 0; ni < NT; ++ni) acc[mi][ni] = vz;

  const bf16* ap[2];
  ap[0] = A + (long)m0 * lda;
  ap[1] = A + (long)(m0 + 16) * lda;
  const bf16* bp[NT];
#pragma unroll
  for (int ni = 0; ni < NT; ++ni) bp[ni] = Bt + (long)(n0 + 16 * ni) * ldb;

  for (int k = 0; k < K; k += 32) {
    v16bf af[2], bfr[NT];
#pragma unroll
    for (int mi = 0; mi < 2; ++mi) af[mi] = load_frag_bf16(ap[mi] + k, lda, lane);
#pragma unroll
    for (int ni = 0; ni < NT; ++ni) bfr[ni] = load_frag_bf16(bp[ni] + k, ldb, lane);
#pragma unroll
    for (int mi = 0; mi < 2; ++mi)
#pragma unroll
      for (int ni = 0; ni < NT; ++ni)
        acc[mi][ni] = wmma_bf16(af[mi], bfr[ni], acc[mi][ni]);
    if (k + 32 < K) { // WGP-scope prefetch of next K-slice
      int r = lane & 15;
      __builtin_prefetch(ap[0] + (long)r * lda + k + 32, 0, 3);
      __builtin_prefetch(ap[1] + (long)r * lda + k + 32, 0, 3);
#pragma unroll
      for (int ni = 0; ni < NT; ++ni)
        __builtin_prefetch(bp[ni] + (long)r * ldb + k + 32, 0, 3);
    }
  }

  // C/D layout: VGPR g holds M = g + 8*(lane>=16), N = lane&15
  int cl = lane & 15, ch = lane >> 4;
#pragma unroll
  for (int mi = 0; mi < 2; ++mi)
#pragma unroll
    for (int ni = 0; ni < NT; ++ni)
#pragma unroll
      for (int g = 0; g < 8; ++g) {
        int rr = m0 + mi * 16 + g + ch * 8;
        C[(long)rr * ldc + n0 + 16 * ni + cl] = acc[mi][ni][g];
      }
}

// ---------------------------------------------------------------------------
// fp32 [R,C] -> bf16 [C,R] tiled transpose (weights, once per call)
// ---------------------------------------------------------------------------
__global__ __launch_bounds__(256) void k_transpose_cvt(
    const float* __restrict__ in, bf16* __restrict__ out, int R, int C) {
  __shared__ float tile[32][33];
  int c0 = blockIdx.x * 32, r0 = blockIdx.y * 32;
  int tx = threadIdx.x, ty = threadIdx.y; // (32,8)
#pragma unroll
  for (int i = 0; i < 32; i += 8)
    tile[ty + i][tx] = in[(long)(r0 + ty + i) * C + c0 + tx];
  __syncthreads();
#pragma unroll
  for (int i = 0; i < 32; i += 8)
    out[(long)(c0 + ty + i) * R + r0 + tx] = (bf16)tile[tx][ty + i];
}

// fp32 -> bf16 copy with per-batch destination remap
__global__ __launch_bounds__(256) void k_cvt_rows(
    const float* __restrict__ src, bf16* __restrict__ dst, int cols,
    int rows_per_b, long dst_bstride) {
  long idx = (long)blockIdx.x * 256 + threadIdx.x;
  long r = idx / cols;
  int  c = (int)(idx % cols);
  int  b = (int)(r / rows_per_b);
  long rr = r % rows_per_b;
  dst[(long)b * dst_bstride + rr * (long)cols + c] = (bf16)src[idx];
}

// ---------------------------------------------------------------------------
// Fused (optional add + bias) -> LayerNorm -> (optional residual), D = 1024
// ---------------------------------------------------------------------------
__global__ __launch_bounds__(256) void k_fused_ln(
    const float* __restrict__ base, const float* __restrict__ add1,
    const float* __restrict__ bias, const float* __restrict__ resid,
    const float* __restrict__ g, const float* __restrict__ bv,
    float* __restrict__ out_f32, bf16* __restrict__ out_bf16,
    int rows_per_b, long bf_bstride) {
  const int D = 1024;
  int r = blockIdx.x, tid = threadIdx.x;
  long ro = (long)r * D;
  float t[4];
#pragma unroll
  for (int i = 0; i < 4; ++i) {
    int c = tid + i * 256;
    float vv = base[ro + c];
    if (add1) vv += add1[ro + c];
    if (bias) vv += bias[c];
    t[i] = vv;
  }
  __shared__ float red[256];
  float s = t[0] + t[1] + t[2] + t[3];
  red[tid] = s; __syncthreads();
  for (int off = 128; off > 0; off >>= 1) {
    if (tid < off) red[tid] += red[tid + off];
    __syncthreads();
  }
  float mu = red[0] * (1.0f / D);
  __syncthreads();
  float vs = 0.f;
#pragma unroll
  for (int i = 0; i < 4; ++i) { float d = t[i] - mu; vs += d * d; }
  red[tid] = vs; __syncthreads();
  for (int off = 128; off > 0; off >>= 1) {
    if (tid < off) red[tid] += red[tid + off];
    __syncthreads();
  }
  float rstd = rsqrtf(red[0] * (1.0f / D) + 1e-5f);

  int b = r / rows_per_b;
  long rr = r % rows_per_b;
  bf16* bp = out_bf16 ? out_bf16 + (long)b * bf_bstride + rr * (long)D : nullptr;
#pragma unroll
  for (int i = 0; i < 4; ++i) {
    int c = tid + i * 256;
    float y = (t[i] - mu) * rstd * g[c] + bv[c];
    if (out_f32) out_f32[ro + c] = (resid ? resid[ro + c] : 0.f) + y;
    if (bp) bp[c] = (bf16)y;
  }
}

// q + u/v broadcast, converted to bf16
__global__ __launch_bounds__(256) void k_add_uv_cvt(
    const float* __restrict__ q, const float* __restrict__ u,
    const float* __restrict__ v, bf16* __restrict__ qu, bf16* __restrict__ qv) {
  long idx = (long)blockIdx.x * 256 + threadIdx.x;
  int c = (int)(idx % DIc);
  float qq = q[idx];
  qu[idx] = (bf16)(qq + u[c]);
  qv[idx] = (bf16)(qq + v[c]);
}

// kv [B,St,2*DI] fp32 -> K [B,H,St,DH] bf16 and V^T [B,H,DH,St] bf16
__global__ __launch_bounds__(256) void k_split_kv(
    const float* __restrict__ kv, bf16* __restrict__ Kb,
    bf16* __restrict__ Vt, int St) {
  long idx = (long)blockIdx.x * 256 + threadIdx.x; // over B*St*DI
  int d = (int)(idx & 63); long t = idx >> 6;
  int h = (int)(t % Hc); t /= Hc;
  int j = (int)(t % St); int b = (int)(t / St);
  const float* row = kv + ((long)b * St + j) * (2 * DIc);
  float kk = row[h * DHc + d];
  float vv = row[DIc + h * DHc + d];
  Kb[(((long)b * Hc + h) * St + j) * DHc + d] = (bf16)kk;
  Vt[(((long)b * Hc + h) * DHc + d) * (long)St + j] = (bf16)vv;
}

// attn = content + rel_shift(pos), causal mask (exact flat-index remap of the
// pad+reshape trick, which interleaves batches). In-place on content.
__global__ __launch_bounds__(256) void k_combine_relshift_mask(
    float* __restrict__ attn, const float* __restrict__ pos) {
  long idx = (long)blockIdx.x * 256 + threadIdx.x; // [B,H,S,St]
  int j = (int)(idx % Stc); long t = idx / Stc;
  int i = (int)(t % Sc); t /= Sc;
  int h = (int)(t % Hc); int b = (int)(t / Hc);

  long F = (((long)b * Sc + i) * Stc + j) * Hc + h;    // (B,S,St,H) flat
  long G = F + (long)Bc * Stc * Hc;                    // skip one (B,St,H) row
  long per_b = (long)(Sc + 1) * Stc * Hc;
  int b2 = (int)(G / per_b);
  long r = G % per_b;
  int ip = (int)(r / ((long)Stc * Hc));
  long r2 = r % ((long)Stc * Hc);
  int j2 = (int)(r2 / Hc);
  int h2 = (int)(r2 % Hc);
  float pv = 0.f;
  if (ip > 0)
    pv = pos[(((long)b2 * Hc + h2) * Sc + (ip - 1)) * (long)Stc + j2];
  float vv = attn[idx] + pv;
  if (j > i + Mc) vv = -1e30f; // tgt_mask formula
  attn[idx] = vv;
}

// softmax over the QUERY axis i (faithful): normalize columns of each (b,h)
// [S,St] matrix. Block = 32 j-lanes x 8 i-slices. Writes bf16 probabilities.
__global__ __launch_bounds__(256) void k_softmax_query_axis(
    const float* __restrict__ X, bf16* __restrict__ P, int S, int St,
    float scale) {
  int bh = blockIdx.y;
  int jj = threadIdx.x & 31, ii = threadIdx.x >> 5;
  int j = blockIdx.x * 32 + jj;
  const float* Xp = X + (long)bh * S * St;
  bf16* Pp = P + (long)bh * S * St;

  __shared__ float sm[8][32];
  __shared__ float ss[8][32];

  float mx = -3.0e38f;
  for (int i = ii; i < S; i += 8)
    mx = fmaxf(mx, Xp[(long)i * St + j] * scale);
  sm[ii][jj] = mx; __syncthreads();
  if (ii == 0) {
    float m = sm[0][jj];
#pragma unroll
    for (int r = 1; r < 8; ++r) m = fmaxf(m, sm[r][jj]);
    sm[0][jj] = m;
  }
  __syncthreads();
  mx = sm[0][jj];

  float sum = 0.f;
  for (int i = ii; i < S; i += 8)
    sum += __expf(Xp[(long)i * St + j] * scale - mx);
  ss[ii][jj] = sum; __syncthreads();
  if (ii == 0) {
    float s = ss[0][jj];
#pragma unroll
    for (int r = 1; r < 8; ++r) s += ss[r][jj];
    ss[0][jj] = s;
  }
  __syncthreads();
  float inv = 1.0f / ss[0][jj];

  for (int i = ii; i < S; i += 8)
    Pp[(long)i * St + j] =
        (bf16)(__expf(Xp[(long)i * St + j] * scale - mx) * inv);
}

// o [B,H,S,DH] fp32 -> [B,S,H*DH] bf16
__global__ __launch_bounds__(256) void k_permute_o(
    const float* __restrict__ o, bf16* __restrict__ ob) {
  long idx = (long)blockIdx.x * 256 + threadIdx.x;
  int d = (int)(idx & 63); long t = idx >> 6;
  int s = (int)(t % Sc); t /= Sc;
  int h = (int)(t % Hc); int b = (int)(t / Hc);
  ob[((long)b * Sc + s) * DIc + h * DHc + d] = (bf16)o[idx];
}

// bias + exact GELU -> bf16
__global__ __launch_bounds__(256) void k_bias_gelu(
    const float* __restrict__ x, const float* __restrict__ bias,
    bf16* __restrict__ y, int cols) {
  long idx = (long)blockIdx.x * 256 + threadIdx.x;
  int c = (int)(idx % cols);
  float t = x[idx] + bias[c];
  y[idx] = (bf16)(0.5f * t * (1.0f + erff(t * 0.70710678118654752f)));
}

// out = out2 + y + b2
__global__ __launch_bounds__(256) void k_final_add(
    const float* __restrict__ out2, const float* __restrict__ y,
    const float* __restrict__ b2, float* __restrict__ out) {
  long idx = (long)blockIdx.x * 256 + threadIdx.x;
  int c = (int)(idx % Dc);
  out[idx] = out2[idx] + y[idx] + b2[c];
}

// ---------------------------------------------------------------------------
// Host side
// ---------------------------------------------------------------------------
static void gemm(hipStream_t s, const bf16* A, const bf16* Bt, float* C,
                 int M, int N, int K, int lda, int ldb, int ldc,
                 long sAb, long sAh, long sBb, long sBh, long sCb, long sCh,
                 int Z, int Hdiv) {
  if (N % 128 == 0) { // wide wave tile: 32x64, 8 WMMAs / 6 frag loads
    dim3 g(M / 128, N / 128, Z);
    k_gemm_nt_bf16<64><<<g, 256, 0, s>>>(A, Bt, C, K, lda, ldb, ldc, sAb, sAh,
                                         sBb, sBh, sCb, sCh, Hdiv);
  } else {            // narrow (N == 64): 32x32 wave tile
    dim3 g(M / 128, N / 64, Z);
    k_gemm_nt_bf16<32><<<g, 256, 0, s>>>(A, Bt, C, K, lda, ldb, ldc, sAb, sAh,
                                         sBb, sBh, sCb, sCh, Hdiv);
  }
}

extern "C" void kernel_launch(void* const* d_in, const int* in_sizes, int n_in,
                              void* d_out, int out_size, void* d_ws,
                              size_t ws_size, hipStream_t stream) {
  (void)in_sizes; (void)n_in; (void)out_size; (void)ws_size;
  const float* x     = (const float*)d_in[0];
  const float* enc   = (const float*)d_in[1];
  const float* pe    = (const float*)d_in[2];
  const float* u     = (const float*)d_in[3];
  const float* v     = (const float*)d_in[4];
  const float* mem   = (const float*)d_in[5];
  /* d_in[6] tgt_mask: reproduced analytically */
  const float* Wq_m  = (const float*)d_in[7];
  const float* Wkv_m = (const float*)d_in[8];
  const float* fcw_m = (const float*)d_in[9];
  const float* fcb_m = (const float*)d_in[10];
  const float* lnm_g = (const float*)d_in[11];
  const float* lnm_b = (const float*)d_in[12];
  const float* Wq_c  = (const float*)d_in[13];
  const float* Wkv_c = (const float*)d_in[14];
  const float* fcw_c = (const float*)d_in[15];
  const float* fcb_c = (const float*)d_in[16];
  const float* lnc_g = (const float*)d_in[17];
  const float* lnc_b = (const float*)d_in[18];
  const float* W1    = (const float*)d_in[19];
  const float* b1    = (const float*)d_in[20];
  const float* W2    = (const float*)d_in[21];
  const float* b2    = (const float*)d_in[22];
  const float* ln1_g = (const float*)d_in[23];
  const float* ln1_b = (const float*)d_in[24];
  const float* ln2_g = (const float*)d_in[25];
  const float* ln2_b = (const float*)d_in[26];
  const float* ln3_g = (const float*)d_in[27];
  const float* ln3_b = (const float*)d_in[28];

  char* ws = (char*)d_ws;
  size_t off = 0;
  auto alloc = [&](size_t n) {
    size_t o = off;
    off += (n + 255) & ~(size_t)255;
    return (void*)(ws + o);
  };

  // Persistent bf16 weights (transposed to [N,K]) + pos-emb
  bf16* WqmT  = (bf16*)alloc((size_t)Dc * DIc * 2);
  bf16* WkvmT = (bf16*)alloc((size_t)Dc * 2 * DIc * 2);
  bf16* fcwmT = (bf16*)alloc((size_t)DIc * Dc * 2);
  bf16* WqcT  = (bf16*)alloc((size_t)Dc * DIc * 2);
  bf16* WkvcT = (bf16*)alloc((size_t)Dc * 2 * DIc * 2);
  bf16* fcwcT = (bf16*)alloc((size_t)DIc * Dc * 2);
  bf16* W1T   = (bf16*)alloc((size_t)Dc * DFFc * 2);
  bf16* W2T   = (bf16*)alloc((size_t)DFFc * Dc * 2);
  bf16* peB   = (bf16*)alloc((size_t)Stc * DIc * 2);
  // Activations (aliased across phases)
  float* bufA  = (float*)alloc((size_t)Bc * Sc * Dc * 4);   // xn / dec
  bf16*  bufB  = (bf16*)alloc((size_t)Bc * Stc * Dc * 2);   // h / decb / h3n
  float* qf    = (float*)alloc((size_t)Bc * Sc * DIc * 4);
  float* kvf   = (float*)alloc((size_t)Bc * Stc * 2 * DIc * 4);
  bf16*  qu    = (bf16*)alloc((size_t)Bc * Sc * DIc * 2);   // also qb (cross)
  bf16*  qv    = (bf16*)alloc((size_t)Bc * Sc * DIc * 2);
  bf16*  Kb    = (bf16*)alloc((size_t)Bc * Hc * Stc * DHc * 2);
  bf16*  Vt    = (bf16*)alloc((size_t)Bc * Hc * DHc * Stc * 2);
  float* of    = (float*)alloc((size_t)Bc * Hc * Sc * DHc * 4);
  bf16*  ob    = (bf16*)alloc((size_t)Bc * Sc * DIc * 2);   // also encb
  float* fco   = (float*)alloc((size_t)Bc * Sc * Dc * 4);   // also FFN y
  float* outb  = (float*)alloc((size_t)Bc * Sc * Dc * 4);
  float* out2b = (float*)alloc((size_t)Bc * Sc * Dc * 4);
  bf16*  g1    = (bf16*)alloc((size_t)Bc * Sc * DFFc * 2);
  float* S1    = (float*)alloc((size_t)Bc * Hc * Sc * Stc * 4); // content/scores/h1
  float* S2f   = (float*)alloc((size_t)Bc * Hc * Sc * Stc * 4); // pos, then P(bf16)
  bf16*  Pb    = (bf16*)S2f;
  bf16*  encb  = ob;

  const float scale = 0.125f; // 1/sqrt(DH)
  auto tr = [&](const float* in, bf16* out, int R, int C) {
    k_transpose_cvt<<<dim3(C / 32, R / 32), dim3(32, 8), 0, stream>>>(in, out, R, C);
  };
  auto blocks = [](long n) { return dim3((unsigned)(n / 256)); };

  // ---- Phase 0: weight prep --------------------------------------------
  tr(Wq_m, WqmT, Dc, DIc);      tr(Wkv_m, WkvmT, Dc, 2 * DIc);
  tr(fcw_m, fcwmT, DIc, Dc);    tr(Wq_c, WqcT, Dc, DIc);
  tr(Wkv_c, WkvcT, Dc, 2 * DIc); tr(fcw_c, fcwcT, DIc, Dc);
  tr(W1, W1T, Dc, DFFc);        tr(W2, W2T, DFFc, Dc);
  k_cvt_rows<<<blocks((long)Stc * DIc), 256, 0, stream>>>(pe, peB, DIc, Stc, 0);
  // mem (raw) into h[:, :M, :]
  k_cvt_rows<<<blocks((long)Bc * Mc * Dc), 256, 0, stream>>>(
      mem, bufB, Dc, Mc, (long)Stc * Dc);

  // ---- Phase 1: self (recurrence) attention ----------------------------
  // xn = LN(x, ln1): fp32 -> bufA, bf16 -> h[:, M:, :]
  k_fused_ln<<<Bc * Sc, 256, 0, stream>>>(x, nullptr, nullptr, nullptr, ln1_g,
                                          ln1_b, bufA, bufB + (size_t)Mc * Dc,
                                          Sc, (long)Stc * Dc);
  // q = xn @ Wq_m (batched over b: xn rows live inside h)
  gemm(stream, bufB + (size_t)Mc * Dc, WqmT, qf, Sc, DIc, Dc, Dc, Dc, DIc,
       (long)Stc * Dc, 0, 0, 0, (long)Sc * DIc, 0, Bc, 1);
  // kv = h @ Wkv_m
  gemm(stream, bufB, WkvmT, kvf, Bc * Stc, 2 * DIc, Dc, Dc, Dc, 2 * DIc,
       0, 0, 0, 0, 0, 0, 1, 1);
  k_add_uv_cvt<<<blocks((long)Bc * Sc * DIc), 256, 0, stream>>>(qf, u, v, qu, qv);
  k_split_kv<<<blocks((long)Bc * Stc * DIc), 256, 0, stream>>>(kvf, Kb, Vt, Stc);
  // content = (q+u) @ k^T  per (b,h)
  gemm(stream, qu, Kb, S1, Sc, Stc, DHc, DIc, DHc, Stc,
       (long)Sc * DIc, DHc, (long)Hc * Stc * DHc, (long)Stc * DHc,
       (long)Hc * Sc * Stc, (long)Sc * Stc, Bc * Hc, Hc);
  // pos = (q+v) @ pe^T  per (b,h)
  gemm(stream, qv, peB, S2f, Sc, Stc, DHc, DIc, DIc, Stc,
       (long)Sc * DIc, DHc, 0, DHc,
       (long)Hc * Sc * Stc, (long)Sc * Stc, Bc * Hc, Hc);
  k_combine_relshift_mask<<<blocks((long)Bc * Hc * Sc * Stc), 256, 0, stream>>>(
      S1, S2f);
  k_softmax_query_axis<<<dim3(Stc / 32, Bc * Hc), 256, 0, stream>>>(
      S1, Pb, Sc, Stc, scale);
  // o = P @ V  per (b,h)
  gemm(stream, Pb, Vt, of, Sc, DHc, Stc, Stc, Stc, DHc,
       (long)Hc * Sc * Stc, (long)Sc * Stc, (long)Hc * DHc * Stc,
       (long)DHc * Stc, (long)Hc * Sc * DHc, (long)Sc * DHc, Bc * Hc, Hc);
  k_permute_o<<<blocks((long)Bc * Hc * Sc * DHc), 256, 0, stream>>>(of, ob);
  gemm(stream, ob, fcwmT, fco, Bc * Sc, Dc, DIc, DIc, DIc, Dc,
       0, 0, 0, 0, 0, 0, 1, 1);
  // out = x + LN(xn + fc(o) + fcb_m, lnm)
  k_fused_ln<<<Bc * Sc, 256, 0, stream>>>(bufA, fco, fcb_m, x, lnm_g, lnm_b,
                                          outb, nullptr, Sc, 0);

  // ---- Phase 2: cross attention ----------------------------------------
  // dec = LN(out, ln2): fp32 -> bufA, bf16 -> bufB
  k_fused_ln<<<Bc * Sc, 256, 0, stream>>>(outb, nullptr, nullptr, nullptr,
                                          ln2_g, ln2_b, bufA, bufB, Sc,
                                          (long)Sc * Dc);
  k_cvt_rows<<<blocks((long)Bc * Sc * Dc), 256, 0, stream>>>(
      enc, encb, Dc, Bc * Sc, 0);
  gemm(stream, bufB, WqcT, qf, Bc * Sc, DIc, Dc, Dc, Dc, DIc,
       0, 0, 0, 0, 0, 0, 1, 1);
  gemm(stream, encb, WkvcT, kvf, Bc * Sc, 2 * DIc, Dc, Dc, Dc, 2 * DIc,
       0, 0, 0, 0, 0, 0, 1, 1);
  k_split_kv<<<blocks((long)Bc * Sc * DIc), 256, 0, stream>>>(kvf, Kb, Vt, Sc);
  k_cvt_rows<<<blocks((long)Bc * Sc * DIc), 256, 0, stream>>>(
      qf, qu, DIc, Bc * Sc, 0);
  // scores = q @ k^T per (b,h)
  gemm(stream, qu, Kb, S1, Sc, Sc, DHc, DIc, DHc, Sc,
       (long)Sc * DIc, DHc, (long)Hc * Sc * DHc, (long)Sc * DHc,
       (long)Hc * Sc * Sc, (long)Sc * Sc, Bc * Hc, Hc);
  k_softmax_query_axis<<<dim3(Sc / 32, Bc * Hc), 256, 0, stream>>>(
      S1, Pb, Sc, Sc, scale);
  gemm(stream, Pb, Vt, of, Sc, DHc, Sc, Sc, Sc, DHc,
       (long)Hc * Sc * Sc, (long)Sc * Sc, (long)Hc * DHc * Sc, (long)DHc * Sc,
       (long)Hc * Sc * DHc, (long)Sc * DHc, Bc * Hc, Hc);
  k_permute_o<<<blocks((long)Bc * Hc * Sc * DHc), 256, 0, stream>>>(of, ob);
  gemm(stream, ob, fcwcT, fco, Bc * Sc, Dc, DIc, DIc, DIc, Dc,
       0, 0, 0, 0, 0, 0, 1, 1);
  // out2 = out + LN(dec + fc(o) + fcb_c, lnc)
  k_fused_ln<<<Bc * Sc, 256, 0, stream>>>(bufA, fco, fcb_c, outb, lnc_g, lnc_b,
                                          out2b, nullptr, Sc, 0);

  // ---- Phase 3: FFN -----------------------------------------------------
  k_fused_ln<<<Bc * Sc, 256, 0, stream>>>(out2b, nullptr, nullptr, nullptr,
                                          ln3_g, ln3_b, nullptr, bufB, Sc,
                                          (long)Sc * Dc);
  gemm(stream, bufB, W1T, S1, Bc * Sc, DFFc, Dc, Dc, Dc, DFFc,
       0, 0, 0, 0, 0, 0, 1, 1);
  k_bias_gelu<<<blocks((long)Bc * Sc * DFFc), 256, 0, stream>>>(S1, b1, g1, DFFc);
  gemm(stream, g1, W2T, fco, Bc * Sc, Dc, DFFc, DFFc, DFFc, Dc,
       0, 0, 0, 0, 0, 0, 1, 1);
  k_final_add<<<blocks((long)Bc * Sc * Dc), 256, 0, stream>>>(
      out2b, fco, b2, (float*)d_out);
}
